// SelfAttention_53944789238449
// MI455X (gfx1250) — matrix operand order repaired and checked by reading, not compiled, not run on hardware
//
#include <hip/hip_runtime.h>

typedef _Float16 v16h __attribute__((ext_vector_type(16)));
typedef _Float16 v8h  __attribute__((ext_vector_type(8)));
typedef float    v8f  __attribute__((ext_vector_type(8)));
typedef float    v4f  __attribute__((ext_vector_type(4)));

constexpr int BB = 2, SS = 2048, HH = 16, DD = 64;
constexpr int BQ = 128;             // queries per workgroup (8 waves x 16)
constexpr int BK = 32;              // keys per KV iteration
// softmax in base-2: fold log2(e) into QK scale and mask constant;
// exp2(t - max t) == exp(s - max s) exactly.
constexpr float LOG2E    = 1.44269504088896340736f;
constexpr float QK_SCALE = 0.125f * LOG2E;
constexpr float NEG_BIG2 = -10000.0f * LOG2E;

constexpr int KH_STRIDE = 72;       // halves per K row in LDS (144B, 16B-aligned rows)
constexpr int VT_STRIDE = 40;       // halves per V^T row in LDS (80B)
constexpr int ROWSTRIDE = HH * DD;  // 1024 floats between seq positions

// half-wave exchange (lane ^ 16) in one VALU op: v_permlanex16_b32 with
// identity nibble selects gathers lane i of the opposite 16-lane row.
__device__ __forceinline__ unsigned xhalf_u(unsigned u) {
    return __builtin_amdgcn_permlanex16(u, u, 0x76543210u, 0xfedcba98u, false, false);
}
__device__ __forceinline__ float xhalf_f(float x) {
    return __builtin_bit_cast(float, xhalf_u(__builtin_bit_cast(unsigned, x)));
}

// ---- one 32-key flash-attention step for one wave (16 queries) ----
template<bool MASK>
__device__ __forceinline__
void compute_tile(const _Float16* __restrict__ sK,
                  const _Float16* __restrict__ sVt,
                  const v16h& bQ0, const v16h& bQ1,
                  v8f (&acc)[4], float& m, float& l,
                  int kb, int qpos, int n, int hi)
{
    union AF { v16h v; v8h h[2]; };

    // load ALL K A-fragments first so the 8 ds_load_b128 can clause and
    // overlap with WMMA issue (partial dscnt waits)
    AF a[4];
    {
        const _Float16* kr0 = &sK[n * KH_STRIDE];
        const _Float16* kr1 = &sK[(16 + n) * KH_STRIDE];
        a[0].h[0] = *(const v8h*)(kr0 + 8 * hi);
        a[0].h[1] = *(const v8h*)(kr0 + 16 + 8 * hi);
        a[1].h[0] = *(const v8h*)(kr0 + 32 + 8 * hi);
        a[1].h[1] = *(const v8h*)(kr0 + 48 + 8 * hi);
        a[2].h[0] = *(const v8h*)(kr1 + 8 * hi);
        a[2].h[1] = *(const v8h*)(kr1 + 16 + 8 * hi);
        a[3].h[0] = *(const v8h*)(kr1 + 32 + 8 * hi);
        a[3].h[1] = *(const v8h*)(kr1 + 48 + 8 * hi);
    }
    v8f st[2];
    {
        v8f z = {0.f, 0.f, 0.f, 0.f, 0.f, 0.f, 0.f, 0.f};
        st[0] = __builtin_amdgcn_wmma_f32_16x16x32_f16(false, a[0].v, false, bQ0,
                                                       (short)0, z, false, false);
        st[1] = __builtin_amdgcn_wmma_f32_16x16x32_f16(false, a[2].v, false, bQ0,
                                                       (short)0, z, false, false);
        st[0] = __builtin_amdgcn_wmma_f32_16x16x32_f16(false, a[1].v, false, bQ1,
                                                       (short)0, st[0], false, false);
        st[1] = __builtin_amdgcn_wmma_f32_16x16x32_f16(false, a[3].v, false, bQ1,
                                                       (short)0, st[1], false, false);
    }

    float smax = -1e30f;
    #pragma unroll
    for (int r = 0; r < 8; ++r) {
        float x0 = st[0][r], x1 = st[1][r];
        if (MASK) {
            const int k0 = kb + r + 8 * hi;
            x0 += (k0 > qpos) ? NEG_BIG2 : 0.0f;
            x1 += (k0 + 16 > qpos) ? NEG_BIG2 : 0.0f;
            st[0][r] = x0; st[1][r] = x1;
        }
        smax = fmaxf(smax, fmaxf(x0, x1));
    }
    smax = fmaxf(smax, xhalf_f(smax));
    const float mNew  = fmaxf(m, smax);
    const float alpha = __builtin_amdgcn_exp2f(m - mNew);

    float lsum = 0.0f;
    unsigned pk[2][4];                    // exp2 weights, f16-pair packed
    #pragma unroll
    for (int t = 0; t < 2; ++t)
        #pragma unroll
        for (int r2 = 0; r2 < 4; ++r2) {
            const float ea = __builtin_amdgcn_exp2f(st[t][2 * r2]     - mNew);
            const float eb = __builtin_amdgcn_exp2f(st[t][2 * r2 + 1] - mNew);
            lsum += ea + eb;
            union { unsigned u; _Float16 h[2]; } p;
            p.h[0] = (_Float16)ea; p.h[1] = (_Float16)eb;
            pk[t][r2] = p.u;
        }
    lsum += xhalf_f(lsum);
    l = l * alpha + lsum;
    m = mNew;

    union BP { v16h v; unsigned u[8]; } bp;
    const bool lo = (hi == 0);
    #pragma unroll
    for (int r2 = 0; r2 < 4; ++r2) {
        const unsigned o0 = xhalf_u(pk[0][r2]);
        const unsigned o1 = xhalf_u(pk[1][r2]);
        bp.u[r2]     = lo ? pk[0][r2] : o1;
        bp.u[4 + r2] = lo ? o0 : pk[1][r2];
    }

    // load ALL V^T A-fragments, then issue the 4 independent PV WMMAs
    AF aV[4];
    #pragma unroll
    for (int c = 0; c < 4; ++c) {
        const _Float16* vr = &sVt[(c * 16 + n) * VT_STRIDE];
        aV[c].h[0] = *(const v8h*)(vr + 16 * hi);
        aV[c].h[1] = *(const v8h*)(vr + 16 * hi + 8);
    }
    #pragma unroll
    for (int c = 0; c < 4; ++c) {
        acc[c] *= alpha;
        acc[c] = __builtin_amdgcn_wmma_f32_16x16x32_f16(false, aV[c].v, false, bp.v,
                                                        (short)0, acc[c], false, false);
    }
}

__global__ __launch_bounds__(256)
void fa_fwd_kernel(const float* __restrict__ Q,
                   const float* __restrict__ K,
                   const float* __restrict__ V,
                   float* __restrict__ O)
{
    // double-buffered tiles: ~19.5 KB of the 320 KB WGP LDS, shared by 8 waves
    __shared__ __attribute__((aligned(16))) _Float16 sK [2][BK * KH_STRIDE]; // [key][dim]
    __shared__ __attribute__((aligned(16))) _Float16 sVt[2][DD * VT_STRIDE]; // [dim][key]

    const int b    = blockIdx.z;
    const int h    = blockIdx.y;
    const int qblk = blockIdx.x;
    const int tid  = threadIdx.x;
    const int wave = tid >> 5;
    const int lane = tid & 31;
    const int n    = lane & 15;
    const int hi   = lane >> 4;

    const int q0   = qblk * BQ + wave * 16;
    const int qpos = q0 + n;
    const size_t base = (size_t)b * SS * HH * DD + (size_t)h * DD;

    // ---- Q^T B-fragments (dims 0..31, 32..63), scale (incl. log2e) folded in ----
    const float* qrow = Q + base + (size_t)qpos * ROWSTRIDE;
    v16h bQ0, bQ1;
    {
        const int d0 = hi * 8;
        v4f f[4], g[4];
        #pragma unroll
        for (int i = 0; i < 4; ++i) {
            f[i] = *(const v4f*)(qrow + d0 + 4 * (i & 1) + 16 * (i >> 1));
            g[i] = *(const v4f*)(qrow + 32 + d0 + 4 * (i & 1) + 16 * (i >> 1));
        }
        #pragma unroll
        for (int i = 0; i < 4; ++i)
            #pragma unroll
            for (int j = 0; j < 4; ++j) {
                bQ0[4 * i + j] = (_Float16)(f[i][j] * QK_SCALE);
                bQ1[4 * i + j] = (_Float16)(g[i][j] * QK_SCALE);
            }
    }

    float m = -1e30f, l = 0.0f;
    v8f acc[4];
    #pragma unroll
    for (int c = 0; c < 4; ++c)
        #pragma unroll
        for (int r = 0; r < 8; ++r) acc[c][r] = 0.0f;

    // ---- cooperative staging maps (256 threads; 8 floats K + 8 floats V each) ----
    const int kRow = tid >> 3, kSeg = tid & 7;     // K: key row, dims kSeg*8..+7
    const int vKp  = tid >> 4, vSeg = tid & 15;    // V: keys {2vKp,2vKp+1}, dims vSeg*4..+3
    const float* kBase = K + base;
    const float* vBase = V + base;

    v4f kReg[2], vReg0, vReg1;                     // register-staged next tile (16 floats)

    auto loadRegs = [&](int it) {
        const int kb = it * BK;
        const float* kr = kBase + (size_t)(kb + kRow) * ROWSTRIDE + kSeg * 8;
        kReg[0] = *(const v4f*)(kr);
        kReg[1] = *(const v4f*)(kr + 4);
        const float* vr = vBase + (size_t)(kb + 2 * vKp) * ROWSTRIDE + vSeg * 4;
        vReg0 = *(const v4f*)(vr);
        vReg1 = *(const v4f*)(vr + ROWSTRIDE);
    };

    auto storeTile = [&](int buf) {
        v8h kh;
        #pragma unroll
        for (int i = 0; i < 2; ++i)
            #pragma unroll
            for (int j = 0; j < 4; ++j)
                kh[4 * i + j] = (_Float16)kReg[i][j];
        *(v8h*)&sK[buf][kRow * KH_STRIDE + kSeg * 8] = kh;
        #pragma unroll
        for (int i = 0; i < 4; ++i) {
            const int dd = (i + vKp + vSeg) & 3;
            const int d  = vSeg * 4 + dd;
            union { _Float16 hh[2]; unsigned u; } pkv;
            pkv.hh[0] = (_Float16)vReg0[dd];
            pkv.hh[1] = (_Float16)vReg1[dd];
            *(unsigned*)&sVt[buf][d * VT_STRIDE + 2 * vKp] = pkv.u;
        }
    };

    const int itLast = (q0 + 15) >> 5;             // only tile needing the causal mask
    const int nIter  = (qblk * BQ + BQ) / BK;      // tiles needed by whole block

    loadRegs(0);
    storeTile(0);
    __syncthreads();

    for (int it = 0; it < nIter; ++it) {
        const int cur = it & 1;
        const bool haveNext = (it + 1 < nIter);
        if (haveNext) loadRegs(it + 1);            // global loads overlap compute below
        if (it < itLast) {
            compute_tile<false>(sK[cur], sVt[cur], bQ0, bQ1, acc, m, l,
                                it * BK, qpos, n, hi);
        } else if (it == itLast) {
            compute_tile<true>(sK[cur], sVt[cur], bQ0, bQ1, acc, m, l,
                               it * BK, qpos, n, hi);
        }
        if (haveNext) storeTile((it + 1) & 1);     // convert + store into other buffer
        __syncthreads();
    }

    // ---- normalize and write O[b, q, h, :] ----
    const float invl = 1.0f / l;
    float* orow = O + base + (size_t)qpos * ROWSTRIDE;
    #pragma unroll
    for (int c = 0; c < 4; ++c) {
        v4f o0, o1;
        #pragma unroll
        for (int r = 0; r < 4; ++r) {
            o0[r] = acc[c][r] * invl;
            o1[r] = acc[c][4 + r] * invl;
        }
        *(v4f*)(orow + c * 16 + 8 * hi)     = o0;
        *(v4f*)(orow + c * 16 + 8 * hi + 4) = o1;
    }
}

extern "C" void kernel_launch(void* const* d_in, const int* in_sizes, int n_in,
                              void* d_out, int out_size, void* d_ws, size_t ws_size,
                              hipStream_t stream) {
    const float* q = (const float*)d_in[0];
    const float* k = (const float*)d_in[1];
    const float* v = (const float*)d_in[2];
    float* o = (float*)d_out;
    dim3 grid(SS / BQ, HH, BB);   // (16, 16, 2)
    fa_fwd_kernel<<<grid, 256, 0, stream>>>(q, k, v, o);
}
